// SAMVisionLayer_10866267259109
// MI455X (gfx1250) — matrix operand order repaired and checked by reading, not compiled, not run on hardware
//
#include <hip/hip_runtime.h>
#include <hip/hip_bf16.h>
#include <cstdint>

// ---------- CDNA5 WMMA plumbing (wave32, 16x16x32 bf16 -> f32) ----------
typedef __attribute__((ext_vector_type(16))) __bf16       v16bf;
typedef __attribute__((ext_vector_type(8)))  float        v8f;
typedef __attribute__((ext_vector_type(4)))  unsigned int v4u;

struct V4x2 { v4u lo, hi; };

__device__ __forceinline__ v4u ld16(const void* p) { return *(const v4u*)p; }

__device__ __forceinline__ v16bf frag(v4u lo, v4u hi) {
  V4x2 t{lo, hi};
  return __builtin_bit_cast(v16bf, t);
}

__device__ __forceinline__ v8f wmma_bf16(v16bf a, v16bf b, v8f c) {
  // (neg_a, A, neg_b, B, c_mod, C, reuse_a, reuse_b)
  return __builtin_amdgcn_wmma_f32_16x16x32_bf16(false, a, false, b, (short)0, c,
                                                 false, false);
}

// CDNA5 async copy: LDS[vDst + off] = MEM[vAddr]; tracked by ASYNCcnt
__device__ __forceinline__ void async_cp_b128(unsigned ldsOff, const void* gaddr) {
  asm volatile("global_load_async_to_lds_b128 %0, %1, off"
               :: "v"(ldsOff), "v"(gaddr)
               : "memory");
}
#define WAIT_ASYNCCNT_0() asm volatile("s_wait_asynccnt 0x0" ::: "memory")
#define WAIT_ASYNCCNT_2() asm volatile("s_wait_asynccnt 0x2" ::: "memory")

__device__ __forceinline__ float waveSum(float v) {
#pragma unroll
  for (int o = 16; o > 0; o >>= 1) v += __shfl_xor(v, o, 32);
  return v;
}

// ---------- weight convert + transpose: dst[N x K] bf16 = src[K x N] f32 ----------
__global__ void cvt_wT_kernel(const float* __restrict__ src, __bf16* __restrict__ dst,
                              int K, int N) {
  long long total = (long long)K * N;
  for (long long i = (long long)blockIdx.x * 256 + threadIdx.x; i < total;
       i += (long long)gridDim.x * 256) {
    int n = (int)(i / K), k = (int)(i % K);
    dst[i] = (__bf16)src[(size_t)k * N + n];
  }
}

// ---------- LayerNorm (+ optional window partition) -> bf16 rows ----------
__global__ __launch_bounds__(256) void ln_kernel(const float* __restrict__ in,
                                                 const float* __restrict__ g,
                                                 const float* __restrict__ b,
                                                 __bf16* __restrict__ out,
                                                 int windowed) {
  const int row = blockIdx.x;
  const int t = threadIdx.x;
  __bf16* orow = out + (size_t)row * 768;
  size_t pix;
  if (windowed) {
    int win = row / 196, tok = row % 196;
    int b_ = win / 25, wy = (win % 25) / 5, wx = win % 5;
    int hh = wy * 14 + tok / 14, ww = wx * 14 + tok % 14;
    if (hh >= 64 || ww >= 64) {  // padded pixel: LN output is exactly zero
      const __bf16 z = (__bf16)0.f;
      orow[t] = z; orow[t + 256] = z; orow[t + 512] = z;
      return;
    }
    pix = ((size_t)(b_ * 64 + hh)) * 64 + ww;
  } else {
    pix = row;
  }
  const float* xr = in + pix * 768;
  float v0 = xr[t], v1 = xr[t + 256], v2 = xr[t + 512];

  __shared__ float red[8];
  __shared__ float sbc;
  float s = waveSum(v0 + v1 + v2);
  if ((t & 31) == 0) red[t >> 5] = s;
  __syncthreads();
  if (t == 0) {
    float m = 0.f;
#pragma unroll
    for (int i = 0; i < 8; ++i) m += red[i];
    sbc = m * (1.f / 768.f);
  }
  __syncthreads();
  float mu = sbc;
  float d0 = v0 - mu, d1 = v1 - mu, d2 = v2 - mu;
  __syncthreads();
  s = waveSum(d0 * d0 + d1 * d1 + d2 * d2);
  if ((t & 31) == 0) red[t >> 5] = s;
  __syncthreads();
  if (t == 0) {
    float m = 0.f;
#pragma unroll
    for (int i = 0; i < 8; ++i) m += red[i];
    sbc = rsqrtf(m * (1.f / 768.f) + 1e-6f);
  }
  __syncthreads();
  float rs = sbc;
  orow[t]       = (__bf16)(d0 * rs * g[t] + b[t]);
  orow[t + 256] = (__bf16)(d1 * rs * g[t + 256] + b[t + 256]);
  orow[t + 512] = (__bf16)(d2 * rs * g[t + 512] + b[t + 512]);
}

// ---------- zero the token-padding rows of Qh (208) and Kh (224) ----------
__global__ void pad_init_kernel(__bf16* __restrict__ qh, __bf16* __restrict__ kh) {
  const long long NQ = 2400LL * 12 * 64;   // Qh rows 196..207
  const long long NK = 2400LL * 28 * 64;   // Kh rows 196..223
  long long i = (long long)blockIdx.x * 256 + threadIdx.x;
  if (i < NQ) {
    int bh = (int)(i / (12 * 64)); int rem = (int)(i % (12 * 64));
    int tok = 196 + rem / 64, d = rem & 63;
    qh[((size_t)bh * 208 + tok) * 64 + d] = (__bf16)0.f;
  } else if (i < NQ + NK) {
    long long j = i - NQ;
    int bh = (int)(j / (28 * 64)); int rem = (int)(j % (28 * 64));
    int tok = 196 + rem / 64, d = rem & 63;
    kh[((size_t)bh * 224 + tok) * 64 + d] = (__bf16)0.f;
  }
}

// ---------- decomposed rel-pos: relh[bh,tok,ky], relw[bh,tok,kx] (f32) ----------
__global__ void relpos_kernel(const __bf16* __restrict__ Qh,
                              const float* __restrict__ rph,
                              const float* __restrict__ rpw,
                              float* __restrict__ relh, float* __restrict__ relw) {
  const long long total = 2400LL * 196 * 14;
  long long i = (long long)blockIdx.x * 256 + threadIdx.x;
  if (i >= total) return;
  int kk = (int)(i % 14);
  long long tt = i / 14;
  int tok = (int)(tt % 196);
  int bh = (int)(tt / 196);
  int iy = tok / 14, ix = tok % 14;
  const __bf16* q = Qh + ((size_t)bh * 208 + tok) * 64;  // unscaled q
  const float* rh = rph + (iy - kk + 13) * 64;
  const float* rw = rpw + (ix - kk + 13) * 64;
  float sh = 0.f, sw = 0.f;
#pragma unroll 8
  for (int c = 0; c < 64; ++c) {
    float qv = (float)q[c];
    sh += qv * rh[c];
    sw += qv * rw[c];
  }
  relh[i] = sh;
  relw[i] = sw;
}

// ---------- WMMA GEMM with async B-tile staging through LDS ----------
// C = A(MxK,bf16) @ Bt(NxK,bf16)^T + bias; fused epilogues:
// mode 0: QKV scatter into Qh/Kh/Vt     mode 1: proj + window-unpartition + residual
// mode 2: GELU(erf) -> bf16             mode 3: residual add -> f32 out
__global__ __launch_bounds__(256) void gemm_bf16_kernel(
    const __bf16* __restrict__ A, const __bf16* __restrict__ Bt,
    const float* __restrict__ bias, int M, int N, int K, int mode,
    const float* __restrict__ res, float* __restrict__ fout,
    __bf16* __restrict__ bfout, __bf16* __restrict__ qh,
    __bf16* __restrict__ kh, __bf16* __restrict__ vt) {
  // double-buffered B tile: 128 cols x 32 K, bf16, [col][32] rows (64 B per col)
  __shared__ __bf16 Bsm[2][128 * 32];

  const int t = threadIdx.x;
  const int lane = t & 31;
  const int wave = t >> 5;
  const int r16 = lane & 15;
  const bool hihalf = lane >= 16;
  const int rowBase = blockIdx.y * 128 + (wave >> 1) * 32;
  const int colBase = blockIdx.x * 128;
  const int waveColOff = (wave & 1) * 64;
  const int aoff = hihalf ? 8 : 0;
  const int boff = hihalf ? 8 : 0;

  // per-thread B copy slice: two 16B chunks
  const int c0col = t >> 2, c0kp = (t & 3) * 8;          // chunk 0
  const int c1col = (t + 256) >> 2, c1kp = ((t + 256) & 3) * 8;  // chunk 1

  const v8f vz = {0.f, 0.f, 0.f, 0.f, 0.f, 0.f, 0.f, 0.f};
  v8f acc[2][4];
#pragma unroll
  for (int mi = 0; mi < 2; ++mi)
#pragma unroll
    for (int ni = 0; ni < 4; ++ni) acc[mi][ni] = vz;

  int arow[2];
#pragma unroll
  for (int mi = 0; mi < 2; ++mi) {
    int r = rowBase + mi * 16 + r16;
    arow[mi] = (r < M) ? r : (M - 1);  // clamp; OOB rows never stored
  }

  const int nk = K / 32;  // K is always a multiple of 32 here

  // prologue: stage k-step 0 into buffer 0 (async; 2 instructions per wave)
  async_cp_b128((unsigned)(uintptr_t)&Bsm[0][c0col * 32 + c0kp],
                Bt + (size_t)(colBase + c0col) * K + c0kp);
  async_cp_b128((unsigned)(uintptr_t)&Bsm[0][c1col * 32 + c1kp],
                Bt + (size_t)(colBase + c1col) * K + c1kp);

  for (int it = 0; it < nk; ++it) {
    const int kb = it * 32;
    __syncthreads();  // all waves finished reading buf[(it+1)&1] last iteration
    if (it + 1 < nk) {
      const int kb2 = kb + 32;
      __bf16* nbuf = Bsm[(it + 1) & 1];
      async_cp_b128((unsigned)(uintptr_t)&nbuf[c0col * 32 + c0kp],
                    Bt + (size_t)(colBase + c0col) * K + kb2 + c0kp);
      async_cp_b128((unsigned)(uintptr_t)&nbuf[c1col * 32 + c1kp],
                    Bt + (size_t)(colBase + c1col) * K + kb2 + c1kp);
      WAIT_ASYNCCNT_2();  // in-order: retires buf[it&1]'s pair, newest 2 in flight
    } else {
      WAIT_ASYNCCNT_0();
    }
    __syncthreads();  // buf[it&1] visible to all waves

    // A fragments direct from global
    v16bf a[2];
#pragma unroll
    for (int mi = 0; mi < 2; ++mi) {
      const __bf16* p = A + (size_t)arow[mi] * K + kb + aoff;
      a[mi] = frag(ld16(p), ld16(p + 16));
    }
    if (kb + 256 < K) {
      __builtin_prefetch(A + (size_t)arow[0] * K + kb + 256, 0, 1);
    }
    // B fragments from LDS
    const __bf16* lb = Bsm[it & 1];
    v16bf b[4];
#pragma unroll
    for (int ni = 0; ni < 4; ++ni) {
      const __bf16* p = lb + (waveColOff + ni * 16 + r16) * 32 + boff;
      b[ni] = frag(ld16(p), ld16(p + 16));
    }
#pragma unroll
    for (int mi = 0; mi < 2; ++mi)
#pragma unroll
      for (int ni = 0; ni < 4; ++ni)
        acc[mi][ni] = wmma_bf16(a[mi], b[ni], acc[mi][ni]);
  }

  // ---- epilogue; hoist div/mod: 4 column-derived + 16 row-derived computations ----
  int cols[4]; float bvals[4]; int csec[4], chh[4], cdd[4];
#pragma unroll
  for (int ni = 0; ni < 4; ++ni) {
    int col = colBase + waveColOff + ni * 16 + r16;
    cols[ni] = col;
    bvals[ni] = bias[col];
    int sec = col / 768, hd = col % 768;
    csec[ni] = sec; chh[ni] = hd >> 6; cdd[ni] = hd & 63;
  }
#pragma unroll
  for (int mi = 0; mi < 2; ++mi) {
#pragma unroll
    for (int v = 0; v < 8; ++v) {
      int row = rowBase + mi * 16 + (hihalf ? 8 : 0) + v;
      if (row >= M) continue;
      int win = 0, tok = 0, hh = 0, ww = 0, b_ = 0;
      bool pixOK = true;
      if (mode == 0 || mode == 1) {
        win = row / 196; tok = row - win * 196;
        if (mode == 1) {
          b_ = win / 25;
          int wrem = win - b_ * 25;
          int wy = wrem / 5, wx = wrem - wy * 5;
          hh = wy * 14 + tok / 14;
          ww = wx * 14 + tok % 14;
          pixOK = (hh < 64) && (ww < 64);
        }
      }
#pragma unroll
      for (int ni = 0; ni < 4; ++ni) {
        float val = acc[mi][ni][v] + bvals[ni];
        if (mode == 0) {
          int bh = win * 12 + chh[ni];
          if (csec[ni] == 0)
            qh[((size_t)bh * 208 + tok) * 64 + cdd[ni]] = (__bf16)val;
          else if (csec[ni] == 1)
            kh[((size_t)bh * 224 + tok) * 64 + cdd[ni]] = (__bf16)val;
          else
            vt[((size_t)bh * 64 + cdd[ni]) * 224 + tok] = (__bf16)val;
        } else if (mode == 1) {
          if (pixOK) {
            size_t idx = (((size_t)(b_ * 64 + hh)) * 64 + ww) * 768 + cols[ni];
            fout[idx] = res[idx] + val;
          }
        } else if (mode == 2) {
          float ge = 0.5f * val * (1.f + erff(val * 0.70710678118f));
          bfout[(size_t)row * N + cols[ni]] = (__bf16)ge;
        } else {
          size_t idx = (size_t)row * 768 + cols[ni];
          fout[idx] = res[idx] + val;
        }
      }
    }
  }
}

// ---------- attention: one block per (window, head); 4 waves, LDS-resident scores ----------
__global__ __launch_bounds__(128) void attn_kernel(
    const __bf16* __restrict__ Qh, const __bf16* __restrict__ Kh,
    const __bf16* __restrict__ Vt, const float* __restrict__ relh,
    const float* __restrict__ relw, __bf16* __restrict__ X2) {
  __shared__ float Ssm[4][16 * 224];  // per-wave 16x224 f32 score strip (57344 B)
  const int bh = blockIdx.x;          // 0..2399
  const int lane = threadIdx.x & 31;
  const int wave = threadIdx.x >> 5;
  const int r16 = lane & 15;
  const bool hihalf = lane >= 16;
  const int aoff = hihalf ? 8 : 0;
  const int boff = hihalf ? 8 : 0;
  float* S = Ssm[wave];
  __bf16* P = (__bf16*)S;  // bf16 probs packed in-place; row stride 448 bf16
  const int win = bh / 12, head = bh % 12;
  const __bf16* Qb = Qh + (size_t)bh * 208 * 64;
  const __bf16* Kb = Kh + (size_t)bh * 224 * 64;
  const __bf16* Vb = Vt + (size_t)bh * 64 * 224;
  const v8f vz = {0.f, 0.f, 0.f, 0.f, 0.f, 0.f, 0.f, 0.f};

  for (int mt = wave; mt < 13; mt += 4) {  // wave-uniform loop: EXEC stays full
    asm volatile("" ::: "memory");
    // ---- S = Q @ K^T (28 WMMAs) ----
    v16bf a0, a1;
    {
      const __bf16* p = Qb + (size_t)(mt * 16 + r16) * 64 + aoff;
      a0 = frag(ld16(p), ld16(p + 16));
      a1 = frag(ld16(p + 32), ld16(p + 48));
    }
    for (int nt = 0; nt < 14; ++nt) {
      const __bf16* p = Kb + (size_t)(nt * 16 + r16) * 64 + boff;
      v16bf b0 = frag(ld16(p), ld16(p + 16));
      v16bf b1 = frag(ld16(p + 32), ld16(p + 48));
      v8f c = vz;
      c = wmma_bf16(a0, b0, c);
      c = wmma_bf16(a1, b1, c);
#pragma unroll
      for (int v = 0; v < 8; ++v)
        S[(v + (hihalf ? 8 : 0)) * 224 + nt * 16 + r16] = c[v];
    }
    asm volatile("" ::: "memory");
    // ---- softmax(scale*S + bias) -> bf16 P ----
    for (int r = 0; r < 16; ++r) {
      int m = mt * 16 + r;
      float vals[7];
      float mx = -3.0e38f;
#pragma unroll
      for (int j = 0; j < 7; ++j) {
        int n = j * 32 + lane;
        float s;
        if (n < 196) {
          s = S[r * 224 + n] * 0.125f;  // 1/sqrt(64)
          if (m < 196) {
            size_t bi = ((size_t)bh * 196 + m) * 14;
            s += relh[bi + n / 14] + relw[bi + n % 14];
          }
        } else {
          s = -3.0e38f;
        }
        vals[j] = s;
        mx = fmaxf(mx, s);
      }
#pragma unroll
      for (int o = 16; o > 0; o >>= 1) mx = fmaxf(mx, __shfl_xor(mx, o, 32));
      float sum = 0.f;
#pragma unroll
      for (int j = 0; j < 7; ++j) {
        float e = (vals[j] > -1.0e37f) ? __expf(vals[j] - mx) : 0.f;
        vals[j] = e;
        sum += e;
      }
#pragma unroll
      for (int o = 16; o > 0; o >>= 1) sum += __shfl_xor(sum, o, 32);
      float inv = 1.f / sum;
      asm volatile("" ::: "memory");
#pragma unroll
      for (int j = 0; j < 7; ++j)
        P[r * 448 + j * 32 + lane] = (__bf16)(vals[j] * inv);
    }
    asm volatile("" ::: "memory");
    // ---- O = P @ V (28 WMMAs; A-fragments from LDS via ds_load_b128) ----
    v8f o[4] = {vz, vz, vz, vz};
    for (int kt = 0; kt < 7; ++kt) {
      int kb = kt * 32;
      const __bf16* pp = P + r16 * 448 + kb + aoff;
      v16bf ap = frag(ld16(pp), ld16(pp + 16));
#pragma unroll
      for (int ni = 0; ni < 4; ++ni) {
        const __bf16* p = Vb + (size_t)(ni * 16 + r16) * 224 + kb + boff;
        v16bf bv = frag(ld16(p), ld16(p + 16));
        o[ni] = wmma_bf16(ap, bv, o[ni]);
      }
    }
#pragma unroll
    for (int ni = 0; ni < 4; ++ni) {
#pragma unroll
      for (int v = 0; v < 8; ++v) {
        int tok = mt * 16 + (hihalf ? 8 : 0) + v;
        if (tok < 196) {
          int col = head * 64 + ni * 16 + r16;
          X2[((size_t)(win * 196 + tok)) * 768 + col] = (__bf16)o[ni][v];
        }
      }
    }
  }
}

// ------------------------------- launcher -------------------------------
extern "C" void kernel_launch(void* const* d_in, const int* in_sizes, int n_in,
                              void* d_out, int out_size, void* d_ws, size_t ws_size,
                              hipStream_t stream) {
  const float* x      = (const float*)d_in[0];
  const float* ln1_g  = (const float*)d_in[1];
  const float* ln1_b  = (const float*)d_in[2];
  const float* qkv_w  = (const float*)d_in[3];
  const float* qkv_b  = (const float*)d_in[4];
  const float* proj_w = (const float*)d_in[5];
  const float* proj_b = (const float*)d_in[6];
  const float* rp_h   = (const float*)d_in[7];
  const float* rp_w   = (const float*)d_in[8];
  const float* ln2_g  = (const float*)d_in[9];
  const float* ln2_b  = (const float*)d_in[10];
  const float* mlp_w1 = (const float*)d_in[11];
  const float* mlp_b1 = (const float*)d_in[12];
  const float* mlp_w2 = (const float*)d_in[13];
  const float* mlp_b2 = (const float*)d_in[14];
  float* out = (float*)d_out;
  (void)in_sizes; (void)n_in; (void)out_size; (void)ws_size;

  char* ws = (char*)d_ws;
  size_t off = 0;
  auto alloc = [&](size_t bytes) -> char* {
    char* p = ws + off;
    off += (bytes + 255) & ~(size_t)255;
    return p;
  };
  __bf16* wqkvT  = (__bf16*)alloc(2304ull * 768 * 2);
  __bf16* wprojT = (__bf16*)alloc(768ull * 768 * 2);
  __bf16* wm1T   = (__bf16*)alloc(3072ull * 768 * 2);
  __bf16* wm2T   = (__bf16*)alloc(768ull * 3072 * 2);
  __bf16* xw     = (__bf16*)alloc(39200ull * 768 * 2);
  __bf16* Qh     = (__bf16*)alloc(2400ull * 208 * 64 * 2);
  __bf16* Kh     = (__bf16*)alloc(2400ull * 224 * 64 * 2);
  __bf16* Vt     = (__bf16*)alloc(2400ull * 64 * 224 * 2);
  float*  relh   = (float*)alloc(2400ull * 196 * 14 * 4);
  float*  relw   = (float*)alloc(2400ull * 196 * 14 * 4);
  __bf16* X2     = (__bf16*)alloc(39200ull * 768 * 2);
  float*  x1     = (float*)alloc(32768ull * 768 * 4);
  __bf16* X3     = (__bf16*)alloc(32768ull * 768 * 2);
  __bf16* Hmid   = (__bf16*)alloc(32768ull * 3072 * 2);

  // 1) weights -> bf16, transposed to [N x K]
  cvt_wT_kernel<<<(2304 * 768 + 255) / 256, 256, 0, stream>>>(qkv_w, wqkvT, 768, 2304);
  cvt_wT_kernel<<<(768 * 768 + 255) / 256, 256, 0, stream>>>(proj_w, wprojT, 768, 768);
  cvt_wT_kernel<<<(3072 * 768 + 255) / 256, 256, 0, stream>>>(mlp_w1, wm1T, 768, 3072);
  cvt_wT_kernel<<<(768 * 3072 + 255) / 256, 256, 0, stream>>>(mlp_w2, wm2T, 3072, 768);

  // 2) LN1 + window partition -> bf16
  ln_kernel<<<39200, 256, 0, stream>>>(x, ln1_g, ln1_b, xw, 1);
  pad_init_kernel<<<(int)((2400LL * 12 * 64 + 2400LL * 28 * 64 + 255) / 256), 256, 0,
                    stream>>>(Qh, Kh);

  // 3) QKV GEMM with Q/K/Vt scatter epilogue
  gemm_bf16_kernel<<<dim3(2304 / 128, (39200 + 127) / 128), 256, 0, stream>>>(
      xw, wqkvT, qkv_b, 39200, 2304, 768, 0, nullptr, nullptr, nullptr, Qh, Kh, Vt);

  // 4) decomposed relative-position terms
  relpos_kernel<<<(int)((2400LL * 196 * 14 + 255) / 256), 256, 0, stream>>>(
      Qh, rp_h, rp_w, relh, relw);

  // 5) attention (WMMA S, softmax+bias in LDS, WMMA PV)
  attn_kernel<<<2400, 128, 0, stream>>>(Qh, Kh, Vt, relh, relw, X2);

  // 6) proj GEMM + window unpartition + residual -> x1 (f32)
  gemm_bf16_kernel<<<dim3(768 / 128, (39200 + 127) / 128), 256, 0, stream>>>(
      X2, wprojT, proj_b, 39200, 768, 768, 1, x, x1, nullptr, nullptr, nullptr, nullptr);

  // 7) LN2 -> bf16
  ln_kernel<<<32768, 256, 0, stream>>>(x1, ln2_g, ln2_b, X3, 0);

  // 8) MLP1 GEMM + exact GELU -> bf16
  gemm_bf16_kernel<<<dim3(3072 / 128, 32768 / 128), 256, 0, stream>>>(
      X3, wm1T, mlp_b1, 32768, 3072, 768, 2, nullptr, nullptr, Hmid, nullptr, nullptr,
      nullptr);

  // 9) MLP2 GEMM + residual -> d_out (f32)
  gemm_bf16_kernel<<<dim3(768 / 128, 32768 / 128), 256, 0, stream>>>(
      Hmid, wm2T, mlp_b2, 32768, 768, 3072, 3, x1, out, nullptr, nullptr, nullptr,
      nullptr);
}
